// CSVAE_1D_13546326851783
// MI455X (gfx1250) — matrix operand order repaired and checked by reading, not compiled, not run on hardware
//
#include <hip/hip_runtime.h>
#include <hip/hip_bf16.h>
#include <math.h>

// ---------------------------------------------------------------------------
// CSVAE forward for MI455X (gfx1250, wave32, WMMA).
//  - NN GEMMs: v_wmma_f32_16x16x32_bf16, block tile 128Mx64N, B panel staged
//    in LDS via async global->LDS B128 copies (ASYNCcnt double-buffered),
//    each wave computes a 16x64 strip (A fragment reused 4x, B from LDS).
//  - 2x linear upsample fused into the consuming GEMM's A fetch.
//  - CovY = AD diag(gamma) AD^H + zeta*I via full-precision
//    V_WMMA_F32_16X16X4_F32 straight into LDS; complex Cholesky, triangular
//    solves and all per-batch reductions fused, one workgroup per batch item.
// ---------------------------------------------------------------------------

typedef __bf16 bf16_t;
typedef __attribute__((ext_vector_type(16))) __bf16 v16bf;
typedef __attribute__((ext_vector_type(8)))  float  v8f;
typedef __attribute__((ext_vector_type(2)))  float  v2f;

#define ODIM 128
#define SDIM 512
#define LDIM 64
#define BS   1024
#define CHOL_S 130   // LDS row stride (floats) to stagger banks

// ---------------------------- tiny kernels ---------------------------------

__global__ void zero_acc_kernel(float* acc) {
  if (threadIdx.x < 4) acc[threadIdx.x] = 0.0f;
}

__global__ void finalize_kernel(const float* __restrict__ acc, float* __restrict__ out) {
  if (threadIdx.x == 0 && blockIdx.x == 0) {
    float kl1 = acc[0] / (float)BS;
    float kl2 = acc[1] / (float)BS;
    float rec = acc[2] / (float)BS;
    out[0] = kl1; out[1] = kl2; out[2] = rec;
    out[3] = -(rec - kl2 - kl1);
  }
}

// z = mu + eps*exp(0.5*log_var)  and KL1 partial sums (atomic into acc[0])
__global__ __launch_bounds__(256) void z_kl1_kernel(
    const float* __restrict__ mu, const float* __restrict__ lv,
    const float* __restrict__ eps, bf16_t* __restrict__ z, float* acc)
{
  __shared__ float red[256];
  int idx = blockIdx.x * 256 + threadIdx.x;  // BS*LDIM = 65536 total
  float m = mu[idx], v = lv[idx], e = eps[idx];
  z[idx] = (bf16_t)(m + e * expf(0.5f * v));
  red[threadIdx.x] = -0.5f * (1.0f + v - m * m - expf(v));
  __syncthreads();
  for (int off = 128; off > 0; off >>= 1) {
    if (threadIdx.x < off) red[threadIdx.x] += red[threadIdx.x + off];
    __syncthreads();
  }
  if (threadIdx.x == 0) atomicAdd(acc + 0, red[0]);
}

// conv4 (768 -> 1 ch) on the on-the-fly upsampled conv3 output, then
// log_gamma = max(.,-10), gamma = exp(log_gamma).
__global__ __launch_bounds__(256) void conv4_gamma_kernel(
    const bf16_t* __restrict__ c3, const float* __restrict__ W4,
    float* __restrict__ gamma)
{
  int idx = blockIdx.x * 256 + threadIdx.x;  // BS*SDIM = 524288 total
  int b = idx >> 9;
  int l = idx & 511;
  int j  = l >> 1;
  int jn = (l & 1) ? ((j + 1 < 256) ? j + 1 : 255) : ((j > 0) ? j - 1 : 0);
  const bf16_t* base = c3 + (long)b * (768L * 256L);
  float s = 0.0f;
  for (int c = 0; c < 768; ++c) {
    float x = 0.75f * (float)base[c * 256 + j] + 0.25f * (float)base[c * 256 + jn];
    s = fmaf(x, W4[c], s);
  }
  gamma[idx] = expf(fmaxf(s, -10.0f));
}

// ------------------------- generic WMMA GEMM -------------------------------
// Out[b, n, l] = act( sum_k A(b,k,l) * B(k,n) + bias[n] )
//   A addressing:  addr = b*aBatch + k*aK + l   (row m = b*L + l)
//   UP=1: A is the 2x linear upsample (half-pixel, edge clamp) of an input of
//         length L/2, fused into the fetch.
//   B: BT=0 -> row-major [K,N] (addr = k*bK + n); BT=1 -> [N,K] (addr = n*bN + k)
//   Out addressing: addr = b*oBatch + n*oN + l
// Block = 256 threads (8 waves), block tile 128M x 64N.  Per 32-K step the
// 32x64 f32 B panel is staged in LDS with async global->LDS B128 copies
// (double buffered on ASYNCcnt); each wave owns a 16M x 64N strip: one A
// fragment (reused 4x) and 4 B fragments built from LDS.
template <typename TA, typename TO, bool UP, bool RELU, bool BT>
__global__ __launch_bounds__(256) void wmma_gemm_kernel(
    const TA* __restrict__ A, const float* __restrict__ B,
    const float* __restrict__ bias, TO* __restrict__ Out,
    int M, int N, int K, int L,
    long aBatch, int aK, int bK, int bN, long oBatch, int oN)
{
  __shared__ float ldsB[2][32 * 64];                 // 16 KB, double buffered

  const int lane = threadIdx.x & 31;
  const int wave = threadIdx.x >> 5;                 // 0..7 -> M sub-tile
  const int half = lane >> 4;
  const int l16  = lane & 15;

  const int tm   = blockIdx.x * 8 + wave;            // 16-row M sub-tile index
  const int tn0g = blockIdx.y * 64;                  // first N column of block

  // ---- async stage of one 32x64 B panel into LDS buffer `buf` -------------
  auto stageB = [&](int k0, int buf) {
    unsigned ldsBase = (unsigned)(uintptr_t)&ldsB[buf][0];
    for (int c = threadIdx.x; c < 512; c += 256) {   // 512 x 16B chunks
      const float* g;
      if (BT) {            // [N,K]: panel stored LDS[n*32 + k]
        int n = c >> 3, kk = (c & 7) * 4;
        g = B + (size_t)(tn0g + n) * bN + (k0 + kk);
      } else {             // [K,N]: panel stored LDS[k*64 + n]
        int k = c >> 4, nn = (c & 15) * 4;
        g = B + (size_t)(k0 + k) * bK + (tn0g + nn);
      }
      unsigned lds = ldsBase + (unsigned)c * 16u;
      unsigned long long ga = (unsigned long long)(uintptr_t)g;
      asm volatile("global_load_async_to_lds_b128 %0, %1, off"
                   :: "v"(lds), "v"(ga) : "memory");
    }
  };

  // ---- A-row addressing (with optional fused upsample) --------------------
  const int arow = tm * 16 + l16;
  long a_off0, a_off1;
  {
    int b = arow / L;
    int l = arow - b * L;
    if (UP) {
      int Lin = L >> 1;
      int j  = l >> 1;
      int jn = (l & 1) ? ((j + 1 < Lin) ? j + 1 : Lin - 1)
                       : ((j > 0) ? j - 1 : 0);
      a_off0 = (long)b * aBatch + j;
      a_off1 = (long)b * aBatch + jn;
    } else {
      a_off0 = (long)b * aBatch + l;
      a_off1 = a_off0;
    }
  }

  v8f acc0 = {}, acc1 = {}, acc2 = {}, acc3 = {};
  stageB(0, 0);

  int buf = 0;
  for (int k0 = 0; k0 < K; k0 += 32, buf ^= 1) {
    const bool issued = (k0 + 32 < K);
    if (issued) stageB(k0 + 32, buf ^ 1);
    // 2 async instructions per wave per stage: allow the just-issued stage to
    // remain outstanding, guarantee the current stage has landed in LDS.
    if (issued) asm volatile("s_wait_asynccnt 0x2" ::: "memory");
    else        asm volatile("s_wait_asynccnt 0x0" ::: "memory");
    __syncthreads();

    // A 16x32 bf16 fragment: lanes 0-15 K={0..15}, lanes 16-31
    // K={16..31} of this 32-step (contiguous, matches B).
    v16bf av;
    const int akb = k0 + half * 16;
#pragma unroll
    for (int j = 0; j < 8; ++j) {
      long kk0 = (long)(akb + j) * aK;
      long kk1 = (long)(akb + 8 + j) * aK;
      float a0, a1;
      if (UP) {
        a0 = 0.75f * (float)A[a_off0 + kk0] + 0.25f * (float)A[a_off1 + kk0];
        a1 = 0.75f * (float)A[a_off0 + kk1] + 0.25f * (float)A[a_off1 + kk1];
      } else {
        a0 = (float)A[a_off0 + kk0];
        a1 = (float)A[a_off0 + kk1];
      }
      av[j]     = (__bf16)a0;
      av[j + 8] = (__bf16)a1;
    }

    // 4 B fragments from the LDS panel; reuse the A fragment 4x.
    const int bkb = half * 16;                        // local K base
#pragma unroll
    for (int nt = 0; nt < 4; ++nt) {
      v16bf bv;
      const int ncol = nt * 16 + l16;                 // local N column
#pragma unroll
      for (int j = 0; j < 16; ++j) {
        float f = BT ? ldsB[buf][ncol * 32 + bkb + j]
                     : ldsB[buf][(bkb + j) * 64 + ncol];
        bv[j] = (__bf16)f;
      }
      v8f* accp = (nt == 0) ? &acc0 : (nt == 1) ? &acc1 : (nt == 2) ? &acc2 : &acc3;
      *accp = __builtin_amdgcn_wmma_f32_16x16x32_bf16(false, av, false, bv,
                                                      (short)0, *accp, false, false);
    }
    __syncthreads();   // protect LDS buffer before next stage overwrites it
  }

  // D layout: element v -> row = v + half*8, col = lane&15 (per sub-tile)
#pragma unroll
  for (int nt = 0; nt < 4; ++nt) {
    v8f accv = (nt == 0) ? acc0 : (nt == 1) ? acc1 : (nt == 2) ? acc2 : acc3;
    int col = tn0g + nt * 16 + l16;
#pragma unroll
    for (int v = 0; v < 8; ++v) {
      int row = tm * 16 + v + half * 8;
      float r = accv[v];
      if (bias) r += bias[col];
      if (RELU) r = fmaxf(r, 0.0f);
      int b = row / L;
      int l = row - b * L;
      Out[(long)b * oBatch + (long)col * oN + l] = (TO)r;
    }
  }
}

// ---------------- fused CovY (WMMA) + Cholesky + solves + reductions -------
__global__ __launch_bounds__(128) void covchol_kernel(
    const float* __restrict__ gamma, const float* __restrict__ ADr,
    const float* __restrict__ ADi,  const float* __restrict__ SI,
    const float* __restrict__ noise_var, float* acc)
{
  extern __shared__ float smem[];
  float* Ar  = smem;                       // 128*CHOL_S
  float* Ai  = Ar + 128 * CHOL_S;          // 128*CHOL_S
  float* yr  = Ai + 128 * CHOL_S;          // 128
  float* yi  = yr + 128;
  float* xr  = yi + 128;
  float* xi  = xr + 128;
  float* pmr = xi + 128;                   // 512
  float* pmi = pmr + 512;
  float* red = pmi + 512;                  // 128

  const int b    = blockIdx.x;
  const int t    = threadIdx.x;
  const int wave = t >> 5;
  const int lane = t & 31;
  const int half = lane >> 4;
  const int l16  = lane & 15;
  const float zeta = noise_var[b];
  const float* gb  = gamma + (long)b * SDIM;

  auto blockReduce = [&](float v) -> float {
    __syncthreads();
    red[t] = v;
    __syncthreads();
    for (int off = 64; off > 0; off >>= 1) {
      if (t < off) red[t] += red[t + off];
      __syncthreads();
    }
    return red[0];
  };

  // ---- Phase 1: CovY tiles (8x8 of 16x16) distributed over the 4 waves ----
  for (int tile = wave; tile < 64; tile += 4) {
    const int tm = tile >> 3, tn = tile & 7;
    const int row = tm * 16 + l16;           // output row (o)
    const int col = tn * 16 + l16;           // output col (o')
    v8f cr = {}, ci = {};
#if __has_builtin(__builtin_amdgcn_wmma_f32_16x16x4_f32)
    // Full-precision f32 WMMA: A 16x4 lanes 0-15 K={k0,k0+1}, lanes 16-31
    // K={k0+2,k0+3}; B 4x16 mirrored.
    for (int k0 = 0; k0 < SDIM; k0 += 4) {
      const int k = k0 + half * 2;
      float g0 = gb[k], g1 = gb[k + 1];
      v2f ar  = {  ADr[row * SDIM + k] * g0,  ADr[row * SDIM + k + 1] * g1 };
      v2f ai  = {  ADi[row * SDIM + k] * g0,  ADi[row * SDIM + k + 1] * g1 };
      v2f nai = { -ai[0], -ai[1] };
      v2f br  = {  ADr[col * SDIM + k],       ADr[col * SDIM + k + 1] };
      v2f bi  = { -ADi[col * SDIM + k],      -ADi[col * SDIM + k + 1] };
      cr = __builtin_amdgcn_wmma_f32_16x16x4_f32(false, ar,  false, br, (short)0, cr, false, false);
      cr = __builtin_amdgcn_wmma_f32_16x16x4_f32(false, nai, false, bi, (short)0, cr, false, false);
      ci = __builtin_amdgcn_wmma_f32_16x16x4_f32(false, ar,  false, bi, (short)0, ci, false, false);
      ci = __builtin_amdgcn_wmma_f32_16x16x4_f32(false, ai,  false, br, (short)0, ci, false, false);
    }
#else
    for (int k0 = 0; k0 < SDIM; k0 += 32) {  // bf16 fallback
      v16bf ar, ai, nai, br, bi;
      const int akb = k0 + half * 8;
#pragma unroll
      for (int j = 0; j < 8; ++j) {
        int ka = akb + j, kc = akb + 16 + j;
        float g0 = gb[ka], g1 = gb[kc];
        float ar0 = ADr[row * SDIM + ka] * g0, ar1 = ADr[row * SDIM + kc] * g1;
        float ai0 = ADi[row * SDIM + ka] * g0, ai1 = ADi[row * SDIM + kc] * g1;
        ar[j] = (__bf16)ar0;   ar[j + 8] = (__bf16)ar1;
        ai[j] = (__bf16)ai0;   ai[j + 8] = (__bf16)ai1;
        nai[j] = (__bf16)(-ai0); nai[j + 8] = (__bf16)(-ai1);
      }
      const int bkb = k0 + half * 16;
#pragma unroll
      for (int j = 0; j < 16; ++j) {
        int kk = bkb + j;
        br[j] = (__bf16)ADr[col * SDIM + kk];
        bi[j] = (__bf16)(-ADi[col * SDIM + kk]);
      }
      cr = __builtin_amdgcn_wmma_f32_16x16x32_bf16(false, ar,  false, br, (short)0, cr, false, false);
      cr = __builtin_amdgcn_wmma_f32_16x16x32_bf16(false, nai, false, bi, (short)0, cr, false, false);
      ci = __builtin_amdgcn_wmma_f32_16x16x32_bf16(false, ar,  false, bi, (short)0, ci, false, false);
      ci = __builtin_amdgcn_wmma_f32_16x16x32_bf16(false, ai,  false, br, (short)0, ci, false, false);
    }
#endif
#pragma unroll
    for (int v = 0; v < 8; ++v) {
      int r = tm * 16 + v + half * 8;
      int c = tn * 16 + l16;
      Ar[r * CHOL_S + c] = cr[v] + ((r == c) ? zeta : 0.0f);
      Ai[r * CHOL_S + c] = ci[v];
    }
  }
  __syncthreads();

  // ---- Phase 2: in-place complex Cholesky (lower), right-looking ----------
  for (int k = 0; k < ODIM; ++k) {
    if (t == 0) {
      float d = sqrtf(Ar[k * CHOL_S + k]);
      Ar[k * CHOL_S + k] = d;
      Ai[k * CHOL_S + k] = 0.0f;
      red[0] = 1.0f / d;
    }
    __syncthreads();
    float inv = red[0];
    if (t > k) { Ar[t * CHOL_S + k] *= inv; Ai[t * CHOL_S + k] *= inv; }
    __syncthreads();
    if (t > k) {
      float lr = Ar[t * CHOL_S + k], li = Ai[t * CHOL_S + k];
      for (int j = k + 1; j <= t; ++j) {          // A[t][j] -= L[t][k]*conj(L[j][k])
        float mr = Ar[j * CHOL_S + k], mi = Ai[j * CHOL_S + k];
        Ar[t * CHOL_S + j] -= lr * mr + li * mi;
        Ai[t * CHOL_S + j] -= li * mr - lr * mi;
      }
    }
    __syncthreads();
  }

  float sumlog = blockReduce(logf(Ar[t * CHOL_S + t]));

  // ---- Phase 3: solve CovY x = Y via L, then L^H --------------------------
  float y0r = SI[(long)b * 256 + t];
  float y0i = SI[(long)b * 256 + 128 + t];
  __syncthreads();
  yr[t] = y0r; yi[t] = y0i;
  __syncthreads();
  for (int j = 0; j < ODIM; ++j) {               // forward: L x1 = Y
    if (t == j) {
      float inv = 1.0f / Ar[j * CHOL_S + j];
      xr[j] = yr[j] * inv; xi[j] = yi[j] * inv;
    }
    __syncthreads();
    if (t > j) {
      float lr = Ar[t * CHOL_S + j], li = Ai[t * CHOL_S + j];
      yr[t] -= lr * xr[j] - li * xi[j];
      yi[t] -= lr * xi[j] + li * xr[j];
    }
    __syncthreads();
  }
  for (int j = ODIM - 1; j >= 0; --j) {          // backward: L^H w = x1
    if (t == j) {
      float inv = 1.0f / Ar[j * CHOL_S + j];
      xr[j] *= inv; xi[j] *= inv;
    }
    __syncthreads();
    if (t < j) {
      float lr = Ar[j * CHOL_S + t], li = Ai[j * CHOL_S + t]; // conj(L[j][t])
      float wr = xr[j], wi = xi[j];
      xr[t] -= lr * wr + li * wi;
      xi[t] -= lr * wi - li * wr;
    }
    __syncthreads();
  }

  // ---- Phase 4: postMeans = gamma .* (AD^H pm); kl2 += |pm|^2/gamma -------
  float kl2p = 0.0f;
  for (int s = t; s < SDIM; s += 128) {
    float pr = 0.0f, pi = 0.0f;
    for (int o = 0; o < ODIM; ++o) {             // conj(AD[o][s]) * pm[o]
      float ar = ADr[o * SDIM + s], ai2 = ADi[o * SDIM + s];
      float wr = xr[o], wi = xi[o];
      pr += ar * wr + ai2 * wi;
      pi += ar * wi - ai2 * wr;
    }
    float g = gb[s];
    pmr[s] = g * pr; pmi[s] = g * pi;
    kl2p += g * (pr * pr + pi * pi);             // |g p|^2 / g
  }
  float kl2sum = blockReduce(kl2p);
  __syncthreads();

  // ---- Phase 5: recon error ----------------------------------------------
  float rr = 0.0f, ri = 0.0f;
  for (int s = 0; s < SDIM; ++s) {
    float ar = ADr[t * SDIM + s], ai2 = ADi[t * SDIM + s];
    rr += pmr[s] * ar - pmi[s] * ai2;
    ri += pmr[s] * ai2 + pmi[s] * ar;
  }
  float er = y0r - rr, ei = y0i - ri;
  float serr = blockReduce(er * er + ei * ei);

  if (t == 0) {
    float kl2_b = -(float)ODIM * logf(zeta) + 2.0f * sumlog + kl2sum;
    float rec_b = -((float)ODIM * logf((float)M_PI * zeta) + serr / zeta);
    atomicAdd(acc + 1, kl2_b);
    atomicAdd(acc + 2, rec_b);
  }
}

// ---------------------------------------------------------------------------

extern "C" void kernel_launch(void* const* d_in, const int* in_sizes, int n_in,
                              void* d_out, int out_size, void* d_ws, size_t ws_size,
                              hipStream_t stream) {
  (void)in_sizes; (void)n_in; (void)out_size; (void)ws_size;
  const float* SI    = (const float*)d_in[0];
  const float* NV    = (const float*)d_in[1];
  const float* EPS   = (const float*)d_in[2];
  const float* ADr   = (const float*)d_in[3];
  const float* ADi   = (const float*)d_in[4];
  const float* encW1 = (const float*)d_in[5];
  const float* encb1 = (const float*)d_in[6];
  const float* encW2 = (const float*)d_in[7];
  const float* encb2 = (const float*)d_in[8];
  const float* encW3 = (const float*)d_in[9];
  const float* encb3 = (const float*)d_in[10];
  const float* muW   = (const float*)d_in[11];
  const float* mub   = (const float*)d_in[12];
  const float* varW  = (const float*)d_in[13];
  const float* varb  = (const float*)d_in[14];
  const float* dec1W = (const float*)d_in[15];
  const float* dec1b = (const float*)d_in[16];
  const float* dec2W = (const float*)d_in[17];
  const float* dec2b = (const float*)d_in[18];
  const float* cv1W  = (const float*)d_in[19];
  const float* cv2W  = (const float*)d_in[20];
  const float* cv3W  = (const float*)d_in[21];
  const float* cv4W  = (const float*)d_in[22];

  char* ws = (char*)d_ws;
  size_t off = 0;
  auto alloc = [&](size_t bytes) -> void* {
    void* p = ws + off;
    off += (bytes + 255) & ~(size_t)255;
    return p;
  };
  float*  acc = (float*)alloc(16);
  bf16_t* h1  = (bf16_t*)alloc(1024L * 256 * 2);
  bf16_t* h2  = (bf16_t*)alloc(1024L * 256 * 2);
  bf16_t* h3  = (bf16_t*)alloc(1024L * 256 * 2);
  float*  mu  = (float*) alloc(1024L * 64 * 4);
  float*  lv  = (float*) alloc(1024L * 64 * 4);
  bf16_t* z   = (bf16_t*)alloc(1024L * 64 * 2);
  bf16_t* dd1 = (bf16_t*)alloc(1024L * 512 * 2);
  bf16_t* dd2 = (bf16_t*)alloc(1024L * 2048 * 2);
  bf16_t* c1  = (bf16_t*)alloc(1024L * 128 * 64 * 2);
  bf16_t* c2  = (bf16_t*)alloc(1024L * 384 * 128 * 2);
  bf16_t* c3  = (bf16_t*)alloc(1024L * 768 * 256 * 2);
  float*  gam = (float*) alloc(1024L * 512 * 4);

  auto grid = [](long M, long N) { return dim3((unsigned)(M / 128), (unsigned)(N / 64)); };

  zero_acc_kernel<<<1, 32, 0, stream>>>(acc);

  // ---------------- encoder ------------------------------------------------
  wmma_gemm_kernel<float, bf16_t, false, true, false><<<grid(1024, 256), 256, 0, stream>>>(
      SI, encW1, encb1, h1, 1024, 256, 256, 1, 256, 1, 256, 1, 256, 1);
  wmma_gemm_kernel<bf16_t, bf16_t, false, true, false><<<grid(1024, 256), 256, 0, stream>>>(
      h1, encW2, encb2, h2, 1024, 256, 256, 1, 256, 1, 256, 1, 256, 1);
  wmma_gemm_kernel<bf16_t, bf16_t, false, true, false><<<grid(1024, 256), 256, 0, stream>>>(
      h2, encW3, encb3, h3, 1024, 256, 256, 1, 256, 1, 256, 1, 256, 1);
  wmma_gemm_kernel<bf16_t, float, false, false, false><<<grid(1024, 64), 256, 0, stream>>>(
      h3, muW, mub, mu, 1024, 64, 256, 1, 256, 1, 64, 1, 64, 1);
  wmma_gemm_kernel<bf16_t, float, false, false, false><<<grid(1024, 64), 256, 0, stream>>>(
      h3, varW, varb, lv, 1024, 64, 256, 1, 256, 1, 64, 1, 64, 1);

  // ---------------- reparameterize + KL1 -----------------------------------
  z_kl1_kernel<<<(1024 * 64) / 256, 256, 0, stream>>>(mu, lv, EPS, z, acc);

  // ---------------- decoder MLP --------------------------------------------
  wmma_gemm_kernel<bf16_t, bf16_t, false, true, false><<<grid(1024, 512), 256, 0, stream>>>(
      z, dec1W, dec1b, dd1, 1024, 512, 64, 1, 64, 1, 512, 1, 512, 1);
  wmma_gemm_kernel<bf16_t, bf16_t, false, true, false><<<grid(1024, 2048), 256, 0, stream>>>(
      dd1, dec2W, dec2b, dd2, 1024, 2048, 512, 1, 512, 1, 2048, 1, 2048, 1);

  // ---------------- 1x1 convs (as GEMMs) with fused upsampling -------------
  wmma_gemm_kernel<bf16_t, bf16_t, false, true, true><<<grid(1024L * 64, 128), 256, 0, stream>>>(
      dd2, cv1W, nullptr, c1,
      1024 * 64, 128, 32, 64, 2048, 64, 1, 32, 128L * 64, 64);
  wmma_gemm_kernel<bf16_t, bf16_t, true, true, true><<<grid(1024L * 128, 384), 256, 0, stream>>>(
      c1, cv2W, nullptr, c2,
      1024 * 128, 384, 128, 128, 128L * 64, 64, 1, 128, 384L * 128, 128);
  wmma_gemm_kernel<bf16_t, bf16_t, true, true, true><<<grid(1024L * 256, 768), 256, 0, stream>>>(
      c2, cv3W, nullptr, c3,
      1024 * 256, 768, 384, 256, 384L * 128, 128, 1, 384, 768L * 256, 256);

  // ---------------- conv4 + gamma ------------------------------------------
  conv4_gamma_kernel<<<(1024 * 512) / 256, 256, 0, stream>>>(c3, cv4W, gam);

  // ---------------- CovY + Cholesky + solves + reductions ------------------
  const int smemBytes = (2 * 128 * CHOL_S + 4 * 128 + 2 * 512 + 128) * (int)sizeof(float);
  covchol_kernel<<<BS, 128, smemBytes, stream>>>(gam, ADr, ADi, SI, NV, acc);

  // ---------------- final 4 scalars ----------------------------------------
  finalize_kernel<<<1, 32, 0, stream>>>(acc, (float*)d_out);
}
